// PointTrans_Layer_up_23673859735700
// MI455X (gfx1250) — matrix operand order repaired and checked by reading, not compiled, not run on hardware
//
#include <hip/hip_runtime.h>

#define B_   4
#define N1_  4096
#define N2_  16384
#define C_   256
#define KUP_ 8

typedef __bf16 bf16_t;
typedef bf16_t v16bf __attribute__((ext_vector_type(16)));
typedef float  v8f   __attribute__((ext_vector_type(8)));
typedef float  v4f   __attribute__((ext_vector_type(4)));

// ---------------------------------------------------------------------------
// Kernel 1: h1 = x1 @ W1^T + b1  via V_WMMA_F32_16X16X32_BF16
//   x1 viewed as [M=16384, K=256] row-major, W1 is [N=256, K=256] row-major,
//   so B-fragment rows of W1 are exactly the columns of W1^T (contiguous K).
//   One 16x16 output tile per wave; 8 K-steps of 32 (fully unrolled -> 8 wmma).
// ---------------------------------------------------------------------------
__global__ void __launch_bounds__(256) gemm_h1_wmma(
    const float* __restrict__ x1, const float* __restrict__ W1,
    const float* __restrict__ b1, float* __restrict__ h1) {
  const int lane = threadIdx.x & 31;
  const int wave = threadIdx.x >> 5;
  const int tile = blockIdx.x * 8 + wave;   // 16384 tiles = (16384/16) * (256/16)
  const int mt   = tile >> 4;               // M-tile index [0,1024)
  const int nt   = tile & 15;               // N-tile index [0,16)
  const int row  = lane & 15;
  const int half = lane >> 4;

  // A fragment source: row (mt*16+row), K chunks {kb..kb+7} and {kb+16..kb+23}, kb=k0+half*8
  const float* aptr = x1 + (size_t)(mt * 16 + row) * C_ + half * 8;
  // B fragment source: column n = nt*16+row of W1^T == row of W1; K chunks {kb..kb+7} and {kb+16..kb+23}, kb=k0+half*8
  const float* bptr = W1 + (size_t)(nt * 16 + row) * C_ + half * 8;

  v8f acc = {};
  #pragma unroll
  for (int k0 = 0; k0 < C_; k0 += 32) {
    v16bf a, b;
    #pragma unroll
    for (int i = 0; i < 8; ++i) {
      a[i]     = (bf16_t)aptr[k0 + i];
      a[8 + i] = (bf16_t)aptr[k0 + 16 + i];
    }
    #pragma unroll
    for (int i = 0; i < 8; ++i) {
      b[i]     = (bf16_t)bptr[k0 + i];
      b[8 + i] = (bf16_t)bptr[k0 + 16 + i];
    }
    acc = __builtin_amdgcn_wmma_f32_16x16x32_bf16(
        /*neg_a=*/false, a, /*neg_b=*/false, b,
        /*c_mod=*/(short)0, acc, /*reuse_a=*/false, /*reuse_b=*/false);
  }

  const float bias = b1[nt * 16 + row];
  // D layout: VGPR v -> row (v + 8*half), col (lane%16)
  float* op = h1 + (size_t)(mt * 16 + half * 8) * C_ + nt * 16 + row;
  #pragma unroll
  for (int v = 0; v < 8; ++v)
    op[(size_t)v * C_] = acc[v] + bias;
}

// ---------------------------------------------------------------------------
// Kernel 2: per-batch KNN (k=8) with inverse-sq-distance weights.
//   pos1[b] (4096 x 3 = 48KB) staged in LDS once per block; one thread/query.
// ---------------------------------------------------------------------------
__global__ void __launch_bounds__(256) knn_topk(
    const float* __restrict__ pos1, const float* __restrict__ pos2,
    int* __restrict__ nidx, float* __restrict__ nwgt) {
  __shared__ float sx[N1_], sy[N1_], sz[N1_];   // 48KB LDS

  const int blocks_per_batch = N2_ / 256;
  const int b    = blockIdx.x / blocks_per_batch;
  const int jblk = blockIdx.x % blocks_per_batch;

  const float* p1 = pos1 + (size_t)b * N1_ * 3;
  for (int i = threadIdx.x; i < N1_; i += 256) {
    sx[i] = p1[i * 3 + 0];
    sy[i] = p1[i * 3 + 1];
    sz[i] = p1[i * 3 + 2];
  }
  __syncthreads();

  const int j = jblk * 256 + threadIdx.x;
  const float* q = pos2 + ((size_t)b * N2_ + j) * 3;
  const float qx = q[0], qy = q[1], qz = q[2];

  float bd[KUP_]; int bi[KUP_];
  #pragma unroll
  for (int k = 0; k < KUP_; ++k) { bd[k] = 3.0e38f; bi[k] = 0; }

  for (int i = 0; i < N1_; ++i) {
    const float dx = qx - sx[i], dy = qy - sy[i], dz = qz - sz[i];
    const float d = dx * dx + dy * dy + dz * dz;
    if (d < bd[KUP_ - 1]) {
      bd[KUP_ - 1] = d; bi[KUP_ - 1] = i;
      #pragma unroll
      for (int t = KUP_ - 1; t > 0; --t) {
        if (bd[t] < bd[t - 1]) {
          float td = bd[t]; bd[t] = bd[t - 1]; bd[t - 1] = td;
          int   ti = bi[t]; bi[t] = bi[t - 1]; bi[t - 1] = ti;
        }
      }
    }
  }

  float w[KUP_], wsum = 0.0f;
  #pragma unroll
  for (int k = 0; k < KUP_; ++k) {
    w[k] = 1.0f / fmaxf(bd[k], 1e-16f);   // matches jnp.clip(d2, EPS)
    wsum += w[k];
  }
  const float inv = 1.0f / wsum;

  const size_t base = ((size_t)b * N2_ + j) * KUP_;
  #pragma unroll
  for (int k = 0; k < KUP_; ++k) {
    nidx[base + k] = bi[k];
    nwgt[base + k] = w[k] * inv;
  }
}

// ---------------------------------------------------------------------------
// Kernel 3: out[b,j,:] = sum_k nwgt[b,j,k] * h1[b, nidx[b,j,k], :]
//   One thread per (query, float4 channel group). h1 is L2-resident (16MB).
// ---------------------------------------------------------------------------
__global__ void __launch_bounds__(256) knn_interp(
    const float* __restrict__ h1, const int* __restrict__ nidx,
    const float* __restrict__ nwgt, float* __restrict__ out) {
  const int t  = blockIdx.x * 256 + threadIdx.x;
  const int c4 = t & 63;          // 64 float4 groups = 256 channels
  const int q  = t >> 6;          // global query in [0, B*N2)
  const int b  = q / N2_;

  const size_t nb = (size_t)q * KUP_;
  const float* hbase = h1 + (size_t)b * N1_ * C_ + c4 * 4;

  v4f acc = {0.0f, 0.0f, 0.0f, 0.0f};
  #pragma unroll
  for (int k = 0; k < KUP_; ++k) {
    const int   id = nidx[nb + k];
    const float w  = nwgt[nb + k];
    const v4f   v  = *(const v4f*)(hbase + (size_t)id * C_);
    acc += w * v;
  }
  *(v4f*)(out + (size_t)q * C_ + c4 * 4) = acc;
}

// ---------------------------------------------------------------------------
extern "C" void kernel_launch(void* const* d_in, const int* in_sizes, int n_in,
                              void* d_out, int out_size, void* d_ws, size_t ws_size,
                              hipStream_t stream) {
  const float* x1   = (const float*)d_in[0];  // [B,N1,Cin]
  const float* pos1 = (const float*)d_in[1];  // [B,N1,3]
  // d_in[2] = x2, d_in[6] = W2, d_in[7] = b2 : dead in the reference output
  const float* pos2 = (const float*)d_in[3];  // [B,N2,3]
  const float* W1   = (const float*)d_in[4];  // [Cout,Cin]
  const float* b1   = (const float*)d_in[5];  // [Cout]
  float* out = (float*)d_out;

  // Workspace layout: h1 (16MB) | nidx (2MB) | nwgt (2MB)
  float* h1   = (float*)d_ws;
  int*   nidx = (int*)(h1 + (size_t)B_ * N1_ * C_);
  float* nwgt = (float*)(nidx + (size_t)B_ * N2_ * KUP_);

  // 1) GEMM: 16384 tiles of 16x16, 8 waves (tiles) per 256-thread block
  gemm_h1_wmma<<<(B_ * N1_ / 16) * (C_ / 16) / 8, 256, 0, stream>>>(x1, W1, b1, h1);

  // 2) KNN: one block per 256 queries of one batch
  knn_topk<<<B_ * (N2_ / 256), 256, 0, stream>>>(pos1, pos2, nidx, nwgt);

  // 3) Interpolate: B*N2*64 threads
  knn_interp<<<(B_ * N2_ * 64) / 256, 256, 0, stream>>>(h1, nidx, nwgt, out);
}
